// AttentionLayer_78632261255284
// MI455X (gfx1250) — 8-bit matrix operand order repaired and checked by reading, compiled, not run on hardware
//
#include <hip/hip_runtime.h>

#define NN    4096
#define EMBED 512
#define HEADS 8
#define HDIM  64
#define HD    512

typedef __attribute__((ext_vector_type(2))) float v2f;
typedef __attribute__((ext_vector_type(8))) float v8f;
typedef __attribute__((ext_vector_type(8))) int   v8i;

__device__ __forceinline__ v8f wmma_f32x4(v2f a, v2f b, v8f c) {
  return __builtin_amdgcn_wmma_f32_16x16x4_f32(false, a, false, b, (short)0, c, false, false);
}
__device__ __forceinline__ v8i wmma_iu8(v8i a, v8i b, v8i c) {
  return __builtin_amdgcn_wmma_i32_16x16x64_iu8(false, a, false, b, c, false, false);
}

__global__ void init_graph_kernel(unsigned char* __restrict__ base,
                                  unsigned char* __restrict__ baseT,
                                  float* __restrict__ factors) {
  int idx = blockIdx.x * blockDim.x + threadIdx.x;
  int i = idx >> 12;
  int j = idx & (NN - 1);
  unsigned char v = (i == j || i == NN - 1 || j == NN - 1) ? 1 : 0;
  base[idx]  = v;
  baseT[idx] = v;
  factors[idx] = (i == j) ? 1.0f : 0.0f;
}

__global__ void scatter_edges_kernel(const int* __restrict__ ei,
                                     unsigned char* __restrict__ base,
                                     unsigned char* __restrict__ baseT, int E) {
  int e = blockIdx.x * blockDim.x + threadIdx.x;
  if (e < E) {
    int s = ei[e];
    int d = ei[E + e];
    base[s * NN + d]  = 1;
    baseT[d * NN + s] = 1;
  }
}

__global__ void log_inplace_kernel(float* __restrict__ f) {
  int idx = blockIdx.x * blockDim.x + threadIdx.x;
  f[idx] = logf(f[idx]);
}

template <int FIRST, int LAST, int WPLANES>
__global__ __launch_bounds__(256) void gemm_iu8_plane(
    const unsigned char* __restrict__ A, const unsigned char* __restrict__ BT,
    float* __restrict__ Cf, float pscale, float fscale,
    float* __restrict__ factors,
    unsigned char* __restrict__ p0, unsigned char* __restrict__ p1,
    unsigned char* __restrict__ p2) {
  const int lane = threadIdx.x & 31;
  const int wave = threadIdx.x >> 5;
  const int wrow = wave >> 2;
  const int wcol = wave & 3;
  const int r0 = blockIdx.y * 128 + wrow * 64;
  const int c0 = blockIdx.x * 256 + wcol * 64;
  const int lm = lane & 15;
  const int lh = lane >> 4;

  v8i acc[4][4];
  v8i iz = {};
#pragma unroll
  for (int rt = 0; rt < 4; ++rt)
#pragma unroll
    for (int ct = 0; ct < 4; ++ct) acc[rt][ct] = iz;

  const unsigned char* arow[4];
  const unsigned char* brow[4];
#pragma unroll
  for (int rt = 0; rt < 4; ++rt)
    arow[rt] = A + (size_t)(r0 + rt * 16 + lm) * NN + 16 * lh;
#pragma unroll
  for (int ct = 0; ct < 4; ++ct)
    brow[ct] = BT + (size_t)(c0 + ct * 16 + lm) * NN + 16 * lh;

  for (int k = 0; k < NN; k += 64) {
    v8i a[4], b[4];
#pragma unroll
    for (int rt = 0; rt < 4; ++rt) {
      union { v8i v; unsigned long long q[4]; } u;
      const unsigned char* p = arow[rt] + k;
      u.q[0] = *(const unsigned long long*)(p);
      u.q[1] = *(const unsigned long long*)(p + 8);
      u.q[2] = *(const unsigned long long*)(p + 32);
      u.q[3] = *(const unsigned long long*)(p + 40);
      a[rt] = u.v;
    }
#pragma unroll
    for (int ct = 0; ct < 4; ++ct) {
      union { v8i v; int4 h[2]; } u;
      const unsigned char* p = brow[ct] + k;
      u.h[0] = *(const int4*)(p);
      u.h[1] = *(const int4*)(p + 32);
      b[ct] = u.v;
    }
#pragma unroll
    for (int rt = 0; rt < 4; ++rt)
#pragma unroll
      for (int ct = 0; ct < 4; ++ct)
        acc[rt][ct] = wmma_iu8(a[rt], b[ct], acc[rt][ct]);
  }

#pragma unroll
  for (int rt = 0; rt < 4; ++rt) {
#pragma unroll
    for (int ct = 0; ct < 4; ++ct) {
#pragma unroll
      for (int i = 0; i < 8; ++i) {
        int m = r0 + rt * 16 + i + 8 * lh;
        int n = c0 + ct * 16 + lm;
        size_t off = (size_t)m * NN + n;
        float t = pscale * (float)acc[rt][ct][i];
        if (!FIRST) t += Cf[off];
        if (!LAST) {
          Cf[off] = t;
        } else {
          factors[off] = fmaxf(factors[off], fscale * t);
          if (WPLANES >= 1) {
            int ti = (int)(t + 0.5f);
            p0[off] = (unsigned char)(ti & 255);
            if (WPLANES >= 2) p1[off] = (unsigned char)((ti >> 8) & 255);
            if (WPLANES >= 3) p2[off] = (unsigned char)((ti >> 16) & 255);
          }
        }
      }
    }
  }
}

__global__ __launch_bounds__(256) void gemm_f32_bias(
    const float* __restrict__ A, const float* __restrict__ B,
    float* __restrict__ C, int Nn, int K, const float* __restrict__ bias) {
  const int lane = threadIdx.x & 31;
  const int wave = threadIdx.x >> 5;
  const int wrow = wave >> 2;
  const int wcol = wave & 3;
  const int r0 = blockIdx.y * 128 + wrow * 64;
  const int c0 = blockIdx.x * 256 + wcol * 64;
  const int lm = lane & 15;
  const int lh = lane >> 4;

  v8f acc[4][4];
  v8f vzero = {};
#pragma unroll
  for (int rt = 0; rt < 4; ++rt)
#pragma unroll
    for (int ct = 0; ct < 4; ++ct) acc[rt][ct] = vzero;

  for (int k = 0; k < K; k += 4) {
    v2f a[4], b[4];
#pragma unroll
    for (int rt = 0; rt < 4; ++rt) {
      const float* p = A + (size_t)(r0 + rt * 16 + lm) * K + k + 2 * lh;
      a[rt] = *(const v2f*)p;
    }
#pragma unroll
    for (int ct = 0; ct < 4; ++ct) {
      const float* p = B + (size_t)(k + 2 * lh) * Nn + c0 + ct * 16 + lm;
      v2f bv;
      bv.x = p[0];
      bv.y = p[Nn];
      b[ct] = bv;
    }
#pragma unroll
    for (int rt = 0; rt < 4; ++rt)
#pragma unroll
      for (int ct = 0; ct < 4; ++ct)
        acc[rt][ct] = wmma_f32x4(a[rt], b[ct], acc[rt][ct]);
  }

#pragma unroll
  for (int rt = 0; rt < 4; ++rt)
#pragma unroll
    for (int ct = 0; ct < 4; ++ct)
#pragma unroll
      for (int i = 0; i < 8; ++i) {
        int m = r0 + rt * 16 + i + 8 * lh;
        int n = c0 + ct * 16 + lm;
        C[(size_t)m * Nn + n] = acc[rt][ct][i] + bias[n];
      }
}

__global__ __launch_bounds__(128) void flash_attn_kernel(
    const float* __restrict__ Q, const float* __restrict__ Km,
    const float* __restrict__ V, const float* __restrict__ LM,
    float* __restrict__ O) {
  __shared__ float lds[4][16 * 17];
  const int lane = threadIdx.x & 31;
  const int wave = threadIdx.x >> 5;
  const int task = blockIdx.x * 4 + wave;
  const int h  = task >> 8;
  const int rb = task & 255;
  const int lm15 = lane & 15;
  const int lh = lane >> 4;
  const int hoff = h * HDIM;
  const int r0 = rb * 16;
  const float NEG_INF = -__builtin_inff();

  v2f aq[16];
  {
    const float* qrow = Q + (size_t)(r0 + lm15) * HD + hoff + 2 * lh;
#pragma unroll
    for (int kk = 0; kk < 16; ++kk) aq[kk] = *(const v2f*)(qrow + kk * 4);
  }

  v8f o[4];
  v8f vzero = {};
#pragma unroll
  for (int t = 0; t < 4; ++t) o[t] = vzero;
  float mrun[8], lrun[8];
#pragma unroll
  for (int i = 0; i < 8; ++i) { mrun[i] = NEG_INF; lrun[i] = 0.0f; }

  for (int ctile = 0; ctile < NN / 16; ++ctile) {
    const int cbase = ctile * 16;

    v8f s = vzero;
    {
      const float* krow = Km + (size_t)(cbase + lm15) * HD + hoff + 2 * lh;
#pragma unroll
      for (int kk = 0; kk < 16; ++kk) {
        v2f b = *(const v2f*)(krow + kk * 4);
        s = wmma_f32x4(aq[kk], b, s);
      }
    }

    float alpha[8];
#pragma unroll
    for (int i = 0; i < 8; ++i) {
      int m = r0 + i + 8 * lh;
      float lmv = LM[(size_t)m * NN + cbase + lm15];
      float sv = s[i] * 0.125f + lmv;

      float rmax = sv;
      rmax = fmaxf(rmax, __shfl_xor(rmax, 1, 16));
      rmax = fmaxf(rmax, __shfl_xor(rmax, 2, 16));
      rmax = fmaxf(rmax, __shfl_xor(rmax, 4, 16));
      rmax = fmaxf(rmax, __shfl_xor(rmax, 8, 16));

      float mnew = fmaxf(mrun[i], rmax);
      float a = (mnew == mrun[i]) ? 1.0f : __expf(mrun[i] - mnew);
      float pv = (sv == NEG_INF) ? 0.0f : __expf(sv - mnew);

      float rsum = pv;
      rsum += __shfl_xor(rsum, 1, 16);
      rsum += __shfl_xor(rsum, 2, 16);
      rsum += __shfl_xor(rsum, 4, 16);
      rsum += __shfl_xor(rsum, 8, 16);

      lrun[i] = lrun[i] * a + rsum;
      mrun[i] = mnew;
      alpha[i] = a;
      s[i] = pv;
    }
#pragma unroll
    for (int t = 0; t < 4; ++t)
#pragma unroll
      for (int i = 0; i < 8; ++i) o[t][i] *= alpha[i];

    __syncthreads();
#pragma unroll
    for (int i = 0; i < 8; ++i)
      lds[wave][(i + 8 * lh) * 17 + lm15] = s[i];
    __syncthreads();

#pragma unroll
    for (int kk = 0; kk < 4; ++kk) {
      v2f pa;
      pa.x = lds[wave][lm15 * 17 + kk * 4 + 2 * lh];
      pa.y = lds[wave][lm15 * 17 + kk * 4 + 2 * lh + 1];
#pragma unroll
      for (int vt = 0; vt < 4; ++vt) {
        const float* vp = V + (size_t)(cbase + kk * 4 + 2 * lh) * HD + hoff + vt * 16 + lm15;
        v2f bv;
        bv.x = vp[0];
        bv.y = vp[HD];
        o[vt] = wmma_f32x4(pa, bv, o[vt]);
      }
    }
  }

#pragma unroll
  for (int i = 0; i < 8; ++i) {
    float inv = 1.0f / lrun[i];
    int m = r0 + i + 8 * lh;
#pragma unroll
    for (int vt = 0; vt < 4; ++vt)
      O[(size_t)m * HD + hoff + vt * 16 + lm15] = o[vt][i] * inv;
  }
}

extern "C" void kernel_launch(void* const* d_in, const int* in_sizes, int n_in,
                              void* d_out, int out_size, void* d_ws, size_t ws_size,
                              hipStream_t stream) {
  const float* X  = (const float*)d_in[0];
  const int*   ei = (const int*)d_in[1];
  const float* Wq = (const float*)d_in[3];
  const float* bq = (const float*)d_in[4];
  const float* Wk = (const float*)d_in[5];
  const float* bk = (const float*)d_in[6];
  const float* Wv = (const float*)d_in[7];
  const float* bv = (const float*)d_in[8];
  const float* Wo = (const float*)d_in[9];
  const float* bo = (const float*)d_in[10];
  float* out = (float*)d_out;
  const int E = in_sizes[1] / 2;

  char* ws = (char*)d_ws;
  const size_t MB = (size_t)NN * NN;
  unsigned char* base  = (unsigned char*)(ws);
  unsigned char* baseT = (unsigned char*)(ws + MB);
  unsigned char* m1lo  = (unsigned char*)(ws + 2 * MB);
  unsigned char* m1hi  = (unsigned char*)(ws + 3 * MB);
  unsigned char* m2lo  = (unsigned char*)(ws + 4 * MB);
  unsigned char* m2hi  = (unsigned char*)(ws + 5 * MB);
  unsigned char* m2h2  = (unsigned char*)(ws + 6 * MB);
  float* Cf      = (float*)(ws + 7 * MB);
  float* factors = (float*)(ws + 11 * MB);
  float* Qb = (float*)(ws + 15 * MB);
  float* Kb = Qb + (size_t)NN * HD;
  float* Vb = Kb + (size_t)NN * HD;
  float* Ob = Vb + (size_t)NN * HD;

  init_graph_kernel<<<(NN * NN) / 256, 256, 0, stream>>>(base, baseT, factors);
  scatter_edges_kernel<<<(E + 255) / 256, 256, 0, stream>>>(ei, base, baseT, E);

  dim3 gmask(NN / 256, NN / 128);
  gemm_iu8_plane<1, 1, 2><<<gmask, 256, 0, stream>>>(
      base, baseT, Cf, 1.0f, 0.5f, factors, m1lo, m1hi, nullptr);
  gemm_iu8_plane<1, 0, 0><<<gmask, 256, 0, stream>>>(
      m1lo, baseT, Cf, 1.0f, 0.0f, nullptr, nullptr, nullptr, nullptr);
  gemm_iu8_plane<0, 1, 3><<<gmask, 256, 0, stream>>>(
      m1hi, baseT, Cf, 256.0f, 0.25f, factors, m2lo, m2hi, m2h2);
  gemm_iu8_plane<1, 0, 0><<<gmask, 256, 0, stream>>>(
      m2lo, baseT, Cf, 1.0f, 0.0f, nullptr, nullptr, nullptr, nullptr);
  gemm_iu8_plane<0, 0, 0><<<gmask, 256, 0, stream>>>(
      m2hi, baseT, Cf, 256.0f, 0.0f, nullptr, nullptr, nullptr, nullptr);
  gemm_iu8_plane<0, 1, 0><<<gmask, 256, 0, stream>>>(
      m2h2, baseT, Cf, 65536.0f, 0.125f, factors, nullptr, nullptr, nullptr);

  log_inplace_kernel<<<(NN * NN) / 256, 256, 0, stream>>>(factors);

  dim3 gproj(HD / 256, NN / 128);
  gemm_f32_bias<<<gproj, 256, 0, stream>>>(X, Wq, Qb, HD, EMBED, bq);
  gemm_f32_bias<<<gproj, 256, 0, stream>>>(X, Wk, Kb, HD, EMBED, bk);
  gemm_f32_bias<<<gproj, 256, 0, stream>>>(X, Wv, Vb, HD, EMBED, bv);

  flash_attn_kernel<<<(HEADS * (NN / 16)) / 4, 128, 0, stream>>>(Qb, Kb, Vb, factors, Ob);

  dim3 gout(EMBED / 256, NN / 128);
  gemm_f32_bias<<<gout, 256, 0, stream>>>(Ob, Wo, out, EMBED, HD, bo);
}
